// RobustBounds_19499151523840
// MI455X (gfx1250) — compile-verified, matrix operand order checked by reading, not run on hardware
//
#include <hip/hip_runtime.h>
#include <hip/hip_bf16.h>

// CDNA5 / gfx1250 implementation of Wong-Kolter robust bounds.
// Dominant work (52.6 GFLOP batched GEMM) runs on v_wmma_f32_16x16x32_bf16
// with 4x M register blocking (one 64x16 slab per wave, shared scaled-B
// fragment). V0 / V0f are reduced in-register (shuffle + f32 atomics),
// never materialized in memory.

#define IN_F  784
#define H1    1024
#define H2    1024
#define NC    10
#define BATCH 32
#define EPS_F 0.1f

typedef __bf16 bf16_t;
typedef __attribute__((ext_vector_type(16))) __bf16 bf16x16;
typedef __attribute__((ext_vector_type(8)))  __bf16 bf16x8;
typedef __attribute__((ext_vector_type(8)))  float  f32x8;

// ---- WMMA bf16 fragment loader -------------------------------------------
// 16-bit A/B operand layout (ISA 7.12.2): lane L holds row (L&15); its 16
// bf16 K-elements are two contiguous runs of 8:
//   k = kbase + 8*(L>>4) + t        (t = 0..7)
//   k = kbase + 16 + 8*(L>>4) + t   (t = 8..15)
// Both runs are 16B-aligned -> global_load_b128.
__device__ __forceinline__ bf16x16 load_frag(const bf16_t* __restrict__ base,
                                             int row, int ld, int kbase, int lane) {
  const bf16_t* p = base + (size_t)row * ld + kbase + ((lane >> 4) << 3);
  bf16x8 lo = *(const bf16x8*)p;
  bf16x8 hi = *(const bf16x8*)(p + 16);
  return __builtin_shufflevector(lo, hi, 0,1,2,3,4,5,6,7,8,9,10,11,12,13,14,15);
}

__device__ __forceinline__ float half_reduce(float v) {
  // reduce across the 16 lanes of each half-wave (N dimension of C tile)
  v += __shfl_xor(v, 1, 32);
  v += __shfl_xor(v, 2, 32);
  v += __shfl_xor(v, 4, 32);
  v += __shfl_xor(v, 8, 32);
  return v;
}

__device__ __forceinline__ f32x8 wmma_bf16(bf16x16 a, bf16x16 b, f32x8 c) {
  return __builtin_amdgcn_wmma_f32_16x16x32_bf16(false, a, false, b, (short)0, c,
                                                 false, false);
}

// Epilogue: per 16x16 C tile (m = r + 8*half, n = lane&15), accumulate
//   Aacc[m] += sum_n C[m,n]*x[n]   and   Eacc[m] += eps * sum_n |C[m,n]|
__device__ __forceinline__ void tile_reduce_store(f32x8 acc, float xv, int lane,
                                                  float* __restrict__ Aacc,
                                                  float* __restrict__ Eacc) {
  int n = lane & 15, half = lane >> 4;
#pragma unroll
  for (int r = 0; r < 8; ++r) {
    float sa = half_reduce(acc[r] * xv);
    float se = half_reduce(fabsf(acc[r]));
    if (n == 0) {
      int m = r + (half << 3);
      atomicAdd(&Aacc[m], sa);
      atomicAdd(&Eacc[m], EPS_F * se);
    }
  }
}

// ---- K0: weight conversion / transposes to bf16 --------------------------
__global__ void k_convert(const float* __restrict__ W1, const float* __restrict__ W2,
                          bf16_t* __restrict__ W1Tbf, bf16_t* __restrict__ W2bf,
                          bf16_t* __restrict__ W2Tbf) {
  int tid = blockIdx.x * blockDim.x + threadIdx.x;
  if (tid < H2 * H1) {
    float w = W2[tid];
    int j = tid >> 10, i = tid & 1023;
    W2bf[tid] = (bf16_t)w;            // W2bf[j][i]  (A operand, row-major)
    W2Tbf[i * H2 + j] = (bf16_t)w;    // W2Tbf[i][h] (B operand for nu2@W2)
  }
  if (tid < H1 * IN_F) {
    int i = tid / IN_F, kc = tid % IN_F;
    W1Tbf[kc * H1 + i] = (bf16_t)W1[tid];  // W1Tbf[kcol][i] (B operand)
  }
}

// ---- K1: layer-1 preactivation bounds -> d1, l1m -------------------------
__global__ void k_layer1(const float* __restrict__ x, const float* __restrict__ W1,
                         const float* __restrict__ b1,
                         float* __restrict__ d1, bf16_t* __restrict__ d1bf,
                         float* __restrict__ l1m) {
  int tid = blockIdx.x * blockDim.x + threadIdx.x;
  if (tid >= BATCH * H1) return;
  int b = tid >> 10, j = tid & 1023;
  const float* xr = x + b * IN_F;
  const float* wr = W1 + (size_t)j * IN_F;
  float nom = 0.f, aw = 0.f;
  for (int k = 0; k < IN_F; ++k) { float w = wr[k]; nom += xr[k] * w; aw += fabsf(w); }
  float zc = nom + b1[j];
  float zl = zc - EPS_F * aw, zu = zc + EPS_F * aw;
  bool  I  = (zl < 0.f) && (zu > 0.f);
  float d  = (zl >= 0.f) ? 1.f : (I ? zu / (zu - zl) : 0.f);
  d1[tid]  = d;
  d1bf[tid] = (bf16_t)d;
  l1m[tid] = I ? zl : 0.f;
}

// ---- K2: V0[b] = (W2 . diag(d1[b])) @ W1, reduced on the fly -------------
// One wave computes a 64x16 slab of V0[b]: 4 j-tiles reuse one d1-scaled B
// fragment (K = 1024 -> 32 K-steps x 4 WMMAs). Per WMMA: ~3 b128 loads.
// Epilogue: Adot2[b,j] += sum_n V0*x, E2[b,j] += eps*sum_n |V0|.
__global__ void k_gemm2(const float* __restrict__ x,
                        const bf16_t* __restrict__ W2bf,
                        const bf16_t* __restrict__ W1Tbf,
                        const bf16_t* __restrict__ d1bf,
                        float* __restrict__ Adot2, float* __restrict__ E2) {
  const int NT = 49;                 // 784/16 n-tiles
  const int JB = 16;                 // 1024/64 j-blocks
  int wid = (blockIdx.x << 3) + (threadIdx.x >> 5);
  if (wid >= BATCH * JB * NT) return;
  int lane = threadIdx.x & 31;
  int nt = wid % NT;                 // fastest: waves in a block share A rows
  int t  = wid / NT;
  int jb = t % JB;
  int b  = t / JB;
  int jbase = jb << 6;               // 64-row slab
  int nbase = nt << 4;
  int mrow = lane & 15;
  int brow = nbase + mrow;
  f32x8 acc0 = {}, acc1 = {}, acc2 = {}, acc3 = {};
  for (int kt = 0; kt < H1 / 32; ++kt) {
    int kb = kt << 5;
    __builtin_prefetch(W1Tbf + (size_t)brow * H1 + kb + 32, 0, 3);
    bf16x16 w = load_frag(W1Tbf, brow, H1, kb, lane);   // B = W1 col-access via T
    bf16x16 s = load_frag(d1bf,  b,    H1, kb, lane);   // diag(d1[b]) on K
    bf16x16 bb;
#pragma unroll
    for (int e = 0; e < 16; ++e) bb[e] = (bf16_t)((float)w[e] * (float)s[e]);
    bf16x16 a0 = load_frag(W2bf, jbase + mrow,      H1, kb, lane);
    bf16x16 a1 = load_frag(W2bf, jbase + 16 + mrow, H1, kb, lane);
    bf16x16 a2 = load_frag(W2bf, jbase + 32 + mrow, H1, kb, lane);
    bf16x16 a3 = load_frag(W2bf, jbase + 48 + mrow, H1, kb, lane);
    acc0 = wmma_bf16(a0, bb, acc0);
    acc1 = wmma_bf16(a1, bb, acc1);
    acc2 = wmma_bf16(a2, bb, acc2);
    acc3 = wmma_bf16(a3, bb, acc3);
  }
  float xv = x[b * IN_F + nbase + (lane & 15)];
  tile_reduce_store(acc0, xv, lane, Adot2 + b * H2 + jbase,      E2 + b * H2 + jbase);
  tile_reduce_store(acc1, xv, lane, Adot2 + b * H2 + jbase + 16, E2 + b * H2 + jbase + 16);
  tile_reduce_store(acc2, xv, lane, Adot2 + b * H2 + jbase + 32, E2 + b * H2 + jbase + 32);
  tile_reduce_store(acc3, xv, lane, Adot2 + b * H2 + jbase + 48, E2 + b * H2 + jbase + 48);
}

// ---- K2b: nu1-only terms + layer-2 ReLU coefficients ---------------------
__global__ void k_layer2(const float* __restrict__ W2, const float* __restrict__ b1,
                         const float* __restrict__ b2,
                         const float* __restrict__ d1, const float* __restrict__ l1m,
                         const float* __restrict__ Adot2, const float* __restrict__ E2,
                         float* __restrict__ d2, float* __restrict__ l2m) {
  int tid = blockIdx.x * blockDim.x + threadIdx.x;
  if (tid >= BATCH * H2) return;
  int b = tid >> 10, j = tid & 1023;
  const float* w2r = W2 + (size_t)j * H1;
  const float* d1r = d1 + b * H1;
  const float* l1r = l1m + b * H1;
  float nub1 = 0.f, cl = 0.f, cu = 0.f;
  for (int i = 0; i < H1; ++i) {
    float w = w2r[i] * d1r[i];
    nub1 += w * b1[i];
    cl   += fmaxf(-w, 0.f) * l1r[i];
    cu   += fmaxf( w, 0.f) * l1r[i];
  }
  float A  = Adot2[tid] + b2[j] + nub1;
  float E  = E2[tid];
  float zl = A - E + cl;
  float zu = A + E - cu;
  bool  I  = (zl < 0.f) && (zu > 0.f);
  d2[tid]  = (zl >= 0.f) ? 1.f : (I ? zu / (zu - zl) : 0.f);
  l2m[tid] = I ? zl : 0.f;
}

// ---- K3: nu2 = (c @ W3) . d2 (padded to 16 rows), plus tiny dot terms ----
__global__ void k_nu2(const int* __restrict__ y, const float* __restrict__ W3,
                      const float* __restrict__ b2, const float* __restrict__ b3,
                      const float* __restrict__ d2, const float* __restrict__ l2m,
                      bf16_t* __restrict__ nu2bf, float* __restrict__ cb3,
                      float* __restrict__ dotb2, float* __restrict__ cross2f) {
  int tid = blockIdx.x * blockDim.x + threadIdx.x;
  if (tid >= BATCH * 16) return;
  int b = tid >> 4, j = tid & 15;
  bf16_t* row = nu2bf + (size_t)tid * H2;
  if (j < NC) {
    int yb = y[b];
    const float* wy  = W3 + (size_t)yb * H2;
    const float* wj  = W3 + (size_t)j * H2;
    const float* d2r = d2 + b * H2;
    const float* l2r = l2m + b * H2;
    float db2 = 0.f, c2 = 0.f;
    for (int h = 0; h < H2; ++h) {
      float nu = (wy[h] - wj[h]) * d2r[h];
      row[h] = (bf16_t)nu;
      db2 += nu * b2[h];
      c2  += fmaxf(-nu, 0.f) * l2r[h];
    }
    cb3[tid] = b3[yb] - b3[j];
    dotb2[tid] = db2;
    cross2f[tid] = c2;
  } else {
    for (int h = 0; h < H2; ++h) row[h] = (bf16_t)0.f;
    cb3[tid] = 0.f; dotb2[tid] = 0.f; cross2f[tid] = 0.f;
  }
}

// ---- K4: nu1f = (nu2 @ W2) . diag(d1), WMMA, fused cross/b1 terms --------
__global__ void k_gemm4(const float* __restrict__ d1, const float* __restrict__ b1,
                        const float* __restrict__ l1m,
                        const bf16_t* __restrict__ nu2bf,
                        const bf16_t* __restrict__ W2Tbf,
                        bf16_t* __restrict__ nu1fbf,
                        float* __restrict__ dotb1f, float* __restrict__ cross1f) {
  int wid = (blockIdx.x << 3) + (threadIdx.x >> 5);
  if (wid >= BATCH * 64) return;
  int lane = threadIdx.x & 31;
  int it = wid & 63;
  int b  = wid >> 6;
  int ibase = it << 4;
  const bf16_t* A = nu2bf + (size_t)b * 16 * H2;
  int arow = lane & 15;
  int brow = ibase + (lane & 15);
  f32x8 acc = {};
  for (int kt = 0; kt < H2 / 32; ++kt) {
    int kb = kt << 5;
    bf16x16 a  = load_frag(A,     arow, H2, kb, lane);
    bf16x16 bb = load_frag(W2Tbf, brow, H2, kb, lane);
    acc = wmma_bf16(a, bb, acc);
  }
  int n = lane & 15, half = lane >> 4;
  int i = ibase + n;
  float dv = d1[b * H1 + i];
  float bv = b1[i];
  float lv = l1m[b * H1 + i];
#pragma unroll
  for (int r = 0; r < 8; ++r) {
    int m = r + (half << 3);
    float nu = acc[r] * dv;
    nu1fbf[((size_t)b * 16 + m) * H1 + i] = (bf16_t)nu;
    float sd = half_reduce(nu * bv);
    float sc = half_reduce(fmaxf(-nu, 0.f) * lv);
    if (n == 0) {
      atomicAdd(&dotb1f[b * 16 + m], sd);
      atomicAdd(&cross1f[b * 16 + m], sc);
    }
  }
}

// ---- K5: V0f = nu1f @ W1, reduced on the fly (Afdot, Ef) -----------------
__global__ void k_gemm5(const float* __restrict__ x,
                        const bf16_t* __restrict__ nu1fbf,
                        const bf16_t* __restrict__ W1Tbf,
                        float* __restrict__ Afdot, float* __restrict__ Ef) {
  int wid = (blockIdx.x << 3) + (threadIdx.x >> 5);
  if (wid >= BATCH * 49) return;
  int lane = threadIdx.x & 31;
  int nt = wid % 49;
  int b  = wid / 49;
  int nbase = nt << 4;
  const bf16_t* A = nu1fbf + (size_t)b * 16 * H1;
  int arow = lane & 15;
  int brow = nbase + (lane & 15);
  f32x8 acc = {};
  for (int kt = 0; kt < H1 / 32; ++kt) {
    int kb = kt << 5;
    bf16x16 a  = load_frag(A,     arow, H1, kb, lane);
    bf16x16 bb = load_frag(W1Tbf, brow, H1, kb, lane);
    acc = wmma_bf16(a, bb, acc);
  }
  float xv = x[b * IN_F + nbase + (lane & 15)];
  tile_reduce_store(acc, xv, lane, Afdot + b * 16, Ef + b * 16);
}

// ---- K6: final combine -> out = -J ---------------------------------------
__global__ void k_final(const float* __restrict__ Afdot, const float* __restrict__ Ef,
                        const float* __restrict__ cb3, const float* __restrict__ dotb2,
                        const float* __restrict__ dotb1f,
                        const float* __restrict__ cross2f, const float* __restrict__ cross1f,
                        float* __restrict__ out) {
  int tid = blockIdx.x * blockDim.x + threadIdx.x;
  if (tid >= BATCH * NC) return;
  int b = tid / NC, j = tid % NC;
  int k = b * 16 + j;
  float Af = Afdot[k] + cb3[k] + dotb2[k] + dotb1f[k];
  float J  = Af - Ef[k] + cross2f[k] + cross1f[k];
  out[tid] = -J;
}

extern "C" void kernel_launch(void* const* d_in, const int* in_sizes, int n_in,
                              void* d_out, int out_size, void* d_ws, size_t ws_size,
                              hipStream_t stream) {
  (void)in_sizes; (void)n_in; (void)out_size; (void)ws_size;
  const float* x  = (const float*)d_in[0];
  const int*   y  = (const int*)d_in[1];
  const float* W1 = (const float*)d_in[2];
  const float* b1 = (const float*)d_in[3];
  const float* W2 = (const float*)d_in[4];
  const float* b2 = (const float*)d_in[5];
  const float* W3 = (const float*)d_in[6];
  const float* b3 = (const float*)d_in[7];
  float* out = (float*)d_out;

  char* ws = (char*)d_ws;
  size_t off = 0;
  auto alloc = [&](size_t bytes) {
    size_t o = off;
    off += (bytes + 255) & ~(size_t)255;
    return o;
  };
  // zero-initialized accumulator region (must lead the workspace)
  size_t o_Adot2  = alloc((size_t)BATCH * H2 * 4);
  size_t o_E2     = alloc((size_t)BATCH * H2 * 4);
  size_t o_dotb1f = alloc((size_t)BATCH * 16 * 4);
  size_t o_cross1f= alloc((size_t)BATCH * 16 * 4);
  size_t o_Afdot  = alloc((size_t)BATCH * 16 * 4);
  size_t o_Ef     = alloc((size_t)BATCH * 16 * 4);
  size_t accBytes = off;
  // non-accumulator scratch
  size_t o_d1     = alloc((size_t)BATCH * H1 * 4);
  size_t o_l1m    = alloc((size_t)BATCH * H1 * 4);
  size_t o_d2     = alloc((size_t)BATCH * H2 * 4);
  size_t o_l2m    = alloc((size_t)BATCH * H2 * 4);
  size_t o_d1bf   = alloc((size_t)BATCH * H1 * 2);
  size_t o_cb3    = alloc((size_t)BATCH * 16 * 4);
  size_t o_dotb2  = alloc((size_t)BATCH * 16 * 4);
  size_t o_cross2f= alloc((size_t)BATCH * 16 * 4);
  size_t o_W2bf   = alloc((size_t)H2 * H1 * 2);
  size_t o_W2Tbf  = alloc((size_t)H2 * H1 * 2);
  size_t o_W1Tbf  = alloc((size_t)IN_F * H1 * 2);
  size_t o_nu2bf  = alloc((size_t)BATCH * 16 * H2 * 2);
  size_t o_nu1fbf = alloc((size_t)BATCH * 16 * H1 * 2);

  float*  Adot2   = (float*)(ws + o_Adot2);
  float*  E2      = (float*)(ws + o_E2);
  float*  dotb1f  = (float*)(ws + o_dotb1f);
  float*  cross1f = (float*)(ws + o_cross1f);
  float*  Afdot   = (float*)(ws + o_Afdot);
  float*  Ef      = (float*)(ws + o_Ef);
  float*  d1      = (float*)(ws + o_d1);
  float*  l1m     = (float*)(ws + o_l1m);
  float*  d2      = (float*)(ws + o_d2);
  float*  l2m     = (float*)(ws + o_l2m);
  bf16_t* d1bf    = (bf16_t*)(ws + o_d1bf);
  float*  cb3     = (float*)(ws + o_cb3);
  float*  dotb2   = (float*)(ws + o_dotb2);
  float*  cross2f = (float*)(ws + o_cross2f);
  bf16_t* W2bf    = (bf16_t*)(ws + o_W2bf);
  bf16_t* W2Tbf   = (bf16_t*)(ws + o_W2Tbf);
  bf16_t* W1Tbf   = (bf16_t*)(ws + o_W1Tbf);
  bf16_t* nu2bf   = (bf16_t*)(ws + o_nu2bf);
  bf16_t* nu1fbf  = (bf16_t*)(ws + o_nu1fbf);

  hipMemsetAsync(ws, 0, accBytes, stream);

  k_convert<<<(H2 * H1 + 255) / 256, 256, 0, stream>>>(W1, W2, W1Tbf, W2bf, W2Tbf);
  k_layer1<<<(BATCH * H1 + 255) / 256, 256, 0, stream>>>(x, W1, b1, d1, d1bf, l1m);
  k_gemm2<<<(BATCH * 16 * 49 + 7) / 8, 256, 0, stream>>>(x, W2bf, W1Tbf, d1bf, Adot2, E2);
  k_layer2<<<(BATCH * H2 + 255) / 256, 256, 0, stream>>>(W2, b1, b2, d1, l1m, Adot2, E2, d2, l2m);
  k_nu2<<<(BATCH * 16 + 255) / 256, 256, 0, stream>>>(y, W3, b2, b3, d2, l2m, nu2bf, cb3, dotb2, cross2f);
  k_gemm4<<<(BATCH * 64 + 7) / 8, 256, 0, stream>>>(d1, b1, l1m, nu2bf, W2Tbf, nu1fbf, dotb1f, cross1f);
  k_gemm5<<<(BATCH * 49 + 7) / 8, 256, 0, stream>>>(x, nu1fbf, W1Tbf, Afdot, Ef);
  k_final<<<(BATCH * NC + 255) / 256, 256, 0, stream>>>(Afdot, Ef, cb3, dotb2, dotb1f, cross2f, cross1f, out);
}
